// MiTStage_31447750541390
// MI455X (gfx1250) — matrix operand order repaired and checked by reading, not compiled, not run on hardware
//
#include <hip/hip_runtime.h>
#include <math.h>

typedef __attribute__((ext_vector_type(16))) _Float16 v16h;
typedef __attribute__((ext_vector_type(8)))  _Float16 v8h;
typedef __attribute__((ext_vector_type(8)))  float    v8f;

// ---------------- problem constants ----------------
#define KB_    8          // batch
#define KC_    128        // channels
#define KW_    80         // H = W = 80 after patch embed
#define KHW_   6400       // 80*80
#define KTOK_  51200L     // 8*6400
#define KKV_   400        // 20*20 reduced tokens per image
#define KHID_  1024       // MixFFN hidden
#define KDIM_  64         // head dim
#define KHEADS_ 2

// ---------------- WMMA GEMM ----------------
// C[M,N] = A[M,K] * B[K,N] (+bias[N]) (gelu?) (+resid)
// A f16 row-major (lda), B f16 row-major [K][N] (ldb)
struct GemmArgs {
  const _Float16* A; int lda;
  const _Float16* B; int ldb;
  float* Cf; _Float16* Ch; int ldc;
  const float* bias;
  const float* resid;
  int M, N, K;
  int gelu;
};

__global__ __launch_bounds__(128)
void gemm_f16_wmma(GemmArgs g) {
  __shared__ alignas(16) _Float16 As[64][40];  // 64 M x 32 K (+pad)
  __shared__ alignas(16) _Float16 Bs[32][72];  // 32 K x 64 N (+pad)
  const int tid  = threadIdx.x;
  const int lane = tid & 31;
  const int wid  = tid >> 5;
  const int wy   = wid >> 1;     // 0..1 : which 32-row strip
  const int wx   = wid & 1;      // 0..1 : which 32-col strip
  const long m0  = (long)blockIdx.y * 64;
  const long n0  = (long)blockIdx.x * 64;

  // staging coordinates
  const int ar  = tid >> 1;               // A row 0..63
  const int akc = (tid & 1) * 16;         // A k sub-block 0/16
  const int bk  = tid >> 2;               // B k row 0..31
  const int bnc = (tid & 3) * 16;         // B n sub-block 0/16/32/48

  const bool aRowsFast = (m0 + 64 <= g.M);
  const bool bColsFast = (n0 + 64 <= g.N);

  // LDS byte addresses for async global->LDS staging (low 32 bits of the
  // generic pointer are the LDS offset; HW adds LDS_BASE).
  const unsigned ldsA = (unsigned)(size_t)&As[ar][akc];
  const unsigned ldsB = (unsigned)(size_t)&Bs[bk][bnc];

  v8f acc[2][2] = {};

  for (int kb = 0; kb < g.K; kb += 32) {
    const bool kFast = (kb + 32 <= g.K);
    __syncthreads();
    {  // stage A tile: 64 rows x 32 K halves
      const long mr = m0 + ar;
      if (aRowsFast && kFast) {
        const _Float16* src = g.A + mr * (long)g.lda + kb + akc;
        // CDNA5 direct global->LDS copy, ASYNCcnt tracked. The offset
        // immediate is added to BOTH the LDS and global addresses.
        asm volatile(
            "global_load_async_to_lds_b128 %0, %1, off\n"
            "global_load_async_to_lds_b128 %0, %1, off offset:16\n"
            :
            : "v"(ldsA), "v"((unsigned long long)src)
            : "memory");
      } else {
        const long mrc = mr < g.M ? mr : (long)g.M - 1;
        const _Float16* arow = g.A + mrc * (long)g.lda;
#pragma unroll
        for (int i = 0; i < 16; ++i) {
          const int k  = kb + akc + i;
          const int kc = k < g.K ? k : g.K - 1;
          _Float16 v = arow[kc];
          if (mr >= g.M || k >= g.K) v = (_Float16)0.f;
          As[ar][akc + i] = v;
        }
      }
    }
    {  // stage B tile: 32 K rows x 64 N halves (B already [K][N])
      const long kr = (long)kb + bk;
      if (bColsFast && kFast) {
        const _Float16* src = g.B + kr * (long)g.ldb + n0 + bnc;
        asm volatile(
            "global_load_async_to_lds_b128 %0, %1, off\n"
            "global_load_async_to_lds_b128 %0, %1, off offset:16\n"
            :
            : "v"(ldsB), "v"((unsigned long long)src)
            : "memory");
      } else {
        const long krc = kr < g.K ? kr : (long)g.K - 1;
        const _Float16* brow = g.B + krc * (long)g.ldb;
#pragma unroll
        for (int i = 0; i < 16; ++i) {
          const long n  = n0 + bnc + i;
          const long nc = n < g.N ? n : (long)g.N - 1;
          _Float16 v = brow[nc];
          if (kr >= g.K || n >= g.N) v = (_Float16)0.f;
          Bs[bk][bnc + i] = v;
        }
      }
    }
    // drain our async global->LDS transfers, then publish the tile
    asm volatile("s_wait_asynccnt 0x0" ::: "memory");
    __syncthreads();

    union Frag { v16h v; v8h h[2]; };
    Frag a[2], b[2];
    // A frag (16x32): lanes 0-15: M=lane, halves0-7 K=0..7, halves8-15 K=16..23
    //                 lanes16-31: M=lane-16, K ranges +8
    const int am = lane & 15;
    const int kq = (lane >> 4) * 8;
#pragma unroll
    for (int mt = 0; mt < 2; ++mt) {
      const int row = wy * 32 + mt * 16 + am;
      a[mt].h[0] = *(const v8h*)&As[row][kq];
      a[mt].h[1] = *(const v8h*)&As[row][kq + 16];
    }
    // B frag (32x16): N=lane&15, halves0-7 K=kq..kq+7, halves8-15 K=kq+16..kq+23
#pragma unroll
    for (int nt = 0; nt < 2; ++nt) {
      const int col = wx * 32 + nt * 16 + am;
#pragma unroll
      for (int i = 0; i < 8; ++i) {
        b[nt].h[0][i] = Bs[kq + i][col];
        b[nt].h[1][i] = Bs[kq + 16 + i][col];
      }
    }
#pragma unroll
    for (int mt = 0; mt < 2; ++mt)
#pragma unroll
      for (int nt = 0; nt < 2; ++nt)
        acc[mt][nt] = __builtin_amdgcn_wmma_f32_16x16x32_f16(
            false, a[mt].v, false, b[nt].v, (short)0, acc[mt][nt], false, false);
  }

  // C/D layout: VGPR r -> M = r + (lane>=16 ? 8 : 0); N = lane&15
  const int cn    = lane & 15;
  const int rbase = (lane >> 4) * 8;
#pragma unroll
  for (int mt = 0; mt < 2; ++mt) {
#pragma unroll
    for (int nt = 0; nt < 2; ++nt) {
      const long col = n0 + wx * 32 + nt * 16 + cn;
      if (col >= g.N) continue;
#pragma unroll
      for (int r = 0; r < 8; ++r) {
        const long row = m0 + wy * 32 + mt * 16 + rbase + r;
        if (row >= g.M) continue;
        float v = acc[mt][nt][r];
        if (g.bias)  v += g.bias[col];
        if (g.gelu)  v = 0.5f * v * (1.0f + erff(v * 0.70710678118654752f));
        if (g.resid) v += g.resid[row * (long)g.ldc + col];
        const long o = row * (long)g.ldc + col;
        if (g.Cf) g.Cf[o] = v;
        if (g.Ch) g.Ch[o] = (_Float16)v;
      }
    }
  }
}

// ---------------- patch embed: 3x3 s2 p1, 64 -> 128, NCHW in, token-major out
__global__ __launch_bounds__(128)
void patch_embed_k(const float* __restrict__ x, const float* __restrict__ w,
                   const float* __restrict__ bias, float* __restrict__ X) {
  __shared__ float patch[64 * 9];
  const int pix = blockIdx.x;
  const int b   = blockIdx.y;
  const int oy = pix / KW_, ox = pix % KW_;
  for (int i = threadIdx.x; i < 576; i += 128) {
    const int ic = i / 9, kk = i % 9;
    const int iy = oy * 2 - 1 + kk / 3;
    const int ix = ox * 2 - 1 + kk % 3;
    float v = 0.f;
    if (iy >= 0 && iy < 160 && ix >= 0 && ix < 160)
      v = x[(((long)b * 64 + ic) * 160 + iy) * 160 + ix];
    patch[i] = v;
  }
  __syncthreads();
  const int oc = threadIdx.x;
  const float* wrow = w + (long)oc * 576;  // [oc][ic][ky][kx] == patch order
  float acc = bias[oc];
#pragma unroll 8
  for (int i = 0; i < 576; ++i) acc += patch[i] * wrow[i];
  X[((long)b * KHW_ + pix) * KC_ + oc] = acc;
}

// ---------------- channel LayerNorm over 128 ch, f32 in -> f16 out
__global__ __launch_bounds__(128)
void layernorm128_k(const float* __restrict__ X, const float* __restrict__ w,
                    const float* __restrict__ b, _Float16* __restrict__ Y) {
  const long t = blockIdx.x;
  const int  c = threadIdx.x;
  const float v = X[t * KC_ + c];
  float s = v, q = v * v;
  for (int off = 16; off; off >>= 1) {
    s += __shfl_xor(s, off, 32);
    q += __shfl_xor(q, off, 32);
  }
  __shared__ float red[2][4];
  if ((threadIdx.x & 31) == 0) { red[0][threadIdx.x >> 5] = s; red[1][threadIdx.x >> 5] = q; }
  __syncthreads();
  s = red[0][0] + red[0][1] + red[0][2] + red[0][3];
  q = red[1][0] + red[1][1] + red[1][2] + red[1][3];
  const float mean = s * (1.f / KC_);
  float var = q * (1.f / KC_) - mean * mean;
  var = fmaxf(var, 0.f);
  const float y = (v - mean) / (sqrtf(var) + 1e-5f) * w[c] + b[c];  // (x-m)/(std+eps)
  Y[t * KC_ + c] = (_Float16)y;
}

// ---------------- im2col for 4x4/s4 k,v convs: xln tokens -> [B*400, 2048]
__global__ __launch_bounds__(256)
void im2col_kv_k(const _Float16* __restrict__ XLN, _Float16* __restrict__ COL) {
  const long idx = (long)blockIdx.x * 256 + threadIdx.x;
  const long total = (long)KB_ * KKV_ * 256;   // 8-half chunks (2048/8)
  if (idx >= total) return;
  const int chunk = (int)(idx % 256);
  const int pos   = (int)((idx / 256) % KKV_);
  const int b     = (int)(idx / (256L * KKV_));
  const int c0  = (chunk * 8) % KC_;
  const int tap = (chunk * 8) / KC_;           // dy*4+dx
  const int dy = tap >> 2, dx = tap & 3;
  const int y = pos / 20, xq = pos % 20;
  const long src = ((long)b * KHW_ + (long)(4 * y + dy) * KW_ + (4 * xq + dx)) * KC_ + c0;
  const long dst = ((long)b * KKV_ + pos) * 2048 + (long)tap * KC_ + c0;
  *(v8h*)&COL[dst] = *(const v8h*)&XLN[src];
}

// ---------------- weight helpers ----------------
// f32 [N][K] -> f16 [K][N] (convert + transpose)
__global__ __launch_bounds__(256)
void cvt_transpose_k(const float* __restrict__ in, _Float16* __restrict__ out,
                     int N, int K) {
  const long idx = (long)blockIdx.x * 256 + threadIdx.x;
  if (idx >= (long)N * K) return;
  const long k = idx / N;
  const long n = idx % N;
  out[idx] = (_Float16)in[n * K + k];
}

// wk/wv OIHW [128 o][128 i][4][4] f32 -> f16 [kdim=tap*128+i][o] (im2col order, transposed)
__global__ __launch_bounds__(256)
void repack_kv_wT_k(const float* __restrict__ w, _Float16* __restrict__ out) {
  const long idx = (long)blockIdx.x * 256 + threadIdx.x;
  if (idx >= 2048L * 128) return;
  const int o    = (int)(idx % 128);
  const int kdim = (int)(idx / 128);
  const int tap  = kdim / 128;       // dy*4+dx
  const int ic   = kdim % 128;
  out[idx] = (_Float16)w[(long)o * 2048 + (long)ic * 16 + tap];
}

// KT token-major [b][j][c] f16 -> KTt [b][c][j] f16
__global__ __launch_bounds__(256)
void transpose_kv_k(const _Float16* __restrict__ in, _Float16* __restrict__ out) {
  const long idx = (long)blockIdx.x * 256 + threadIdx.x;
  if (idx >= (long)KB_ * KC_ * KKV_) return;
  const int j = (int)(idx % KKV_);
  const int c = (int)((idx / KKV_) % KC_);
  const int b = (int)(idx / ((long)KKV_ * KC_));
  out[idx] = in[((long)b * KKV_ + j) * KC_ + c];
}

// ---------------- row softmax (scaled), f32 in -> f16 probs ----------------
__global__ __launch_bounds__(256)
void softmax_rows_k(const float* __restrict__ S, _Float16* __restrict__ P,
                    int rows, int cols, float scale) {
  const int row = blockIdx.x * 8 + (threadIdx.x >> 5);
  if (row >= rows) return;
  const int lane = threadIdx.x & 31;
  const float* s = S + (long)row * cols;
  float mx = -3.4e38f;
  for (int j = lane; j < cols; j += 32) mx = fmaxf(mx, s[j] * scale);
  for (int off = 16; off; off >>= 1) mx = fmaxf(mx, __shfl_xor(mx, off, 32));
  float sum = 0.f;
  for (int j = lane; j < cols; j += 32) sum += expf(s[j] * scale - mx);
  for (int off = 16; off; off >>= 1) sum += __shfl_xor(sum, off, 32);
  const float inv = 1.f / sum;
  _Float16* p = P + (long)row * cols;
  for (int j = lane; j < cols; j += 32)
    p[j] = (_Float16)(expf(s[j] * scale - mx) * inv);
}

// ---------------- depthwise 3x3 p1 on token-major [B,6400,1024] f16 ----------
__global__ __launch_bounds__(128)
void dwconv3x3_k(const _Float16* __restrict__ H1, const float* __restrict__ wdw,
                 const float* __restrict__ bdw, _Float16* __restrict__ H2) {
  const int pix = blockIdx.x;
  const int b   = blockIdx.y;
  const int c0  = threadIdx.x * 8;
  const int y = pix / KW_, x = pix % KW_;
  float acc[8];
#pragma unroll
  for (int i = 0; i < 8; ++i) acc[i] = bdw[c0 + i];
#pragma unroll
  for (int ky = 0; ky < 3; ++ky) {
    const int yy = y - 1 + ky;
    if (yy < 0 || yy >= KW_) continue;
#pragma unroll
    for (int kx = 0; kx < 3; ++kx) {
      const int xx = x - 1 + kx;
      if (xx < 0 || xx >= KW_) continue;
      const int tap = ky * 3 + kx;
      const v8h v = *(const v8h*)&H1[((long)b * KHW_ + (long)yy * KW_ + xx) * KHID_ + c0];
#pragma unroll
      for (int i = 0; i < 8; ++i)
        acc[i] += (float)v[i] * wdw[(long)(c0 + i) * 9 + tap];
    }
  }
  v8h o;
#pragma unroll
  for (int i = 0; i < 8; ++i) o[i] = (_Float16)acc[i];
  *(v8h*)&H2[((long)b * KHW_ + pix) * KHID_ + c0] = o;
}

// ---------------- token-major -> NCHW output ----------------
__global__ __launch_bounds__(256)
void tok_to_nchw_k(const float* __restrict__ X, float* __restrict__ out, long n) {
  const long i = (long)blockIdx.x * 256 + threadIdx.x;
  if (i >= n) return;
  const int p = (int)(i % KHW_);
  const int c = (int)((i / KHW_) % KC_);
  const int b = (int)(i / ((long)KHW_ * KC_));
  out[i] = X[((long)b * KHW_ + p) * KC_ + c];
}

// =====================================================================
extern "C" void kernel_launch(void* const* d_in, const int* in_sizes, int n_in,
                              void* d_out, int out_size, void* d_ws, size_t ws_size,
                              hipStream_t stream) {
  (void)in_sizes; (void)out_size; (void)ws_size;
  if (n_in < 3 + 4 * 16) return;
  const float* x    = (const float*)d_in[0];
  const float* pe_w = (const float*)d_in[1];
  const float* pe_b = (const float*)d_in[2];

  char* ws = (char*)d_ws;
  size_t off = 0;
  auto alloc = [&](size_t bytes) -> void* {
    void* p = ws + off;
    off = (off + bytes + 255) & ~(size_t)255;
    return p;
  };
  float*    X   = (float*)alloc(KTOK_ * KC_ * sizeof(float));
  _Float16* XLN = (_Float16*)alloc(KTOK_ * KC_ * 2);
  _Float16* Q   = (_Float16*)alloc(KTOK_ * KC_ * 2);
  _Float16* COL = (_Float16*)alloc((long)KB_ * KKV_ * 2048 * 2);
  _Float16* KT  = (_Float16*)alloc((long)KB_ * KKV_ * KC_ * 2);  // [b][j][c]
  _Float16* KTt = (_Float16*)alloc((long)KB_ * KKV_ * KC_ * 2);  // [b][c][j]
  _Float16* VT  = (_Float16*)alloc((long)KB_ * KKV_ * KC_ * 2);  // [b][j][c]
  float*    Sb  = (float*)alloc((long)KHW_ * KKV_ * sizeof(float));   // per (b,h)
  _Float16* Pb  = (_Float16*)alloc((long)KHW_ * KKV_ * 2);            // per (b,h)
  _Float16* AO  = (_Float16*)alloc(KTOK_ * KC_ * 2);
  _Float16* H1  = (_Float16*)alloc(KTOK_ * KHID_ * 2);  // w1 out, reused as pw out
  _Float16* H2  = (_Float16*)alloc(KTOK_ * KHID_ * 2);  // dw out
  _Float16* WQT = (_Float16*)alloc(128L * 128 * 2);     // [K][N] transposed weights
  _Float16* WKT = (_Float16*)alloc(2048L * 128 * 2);
  _Float16* WVT = (_Float16*)alloc(2048L * 128 * 2);
  _Float16* WOT = (_Float16*)alloc(128L * 128 * 2);
  _Float16* W1T = (_Float16*)alloc(128L * 1024 * 2);
  _Float16* WPT = (_Float16*)alloc(1024L * 1024 * 2);
  _Float16* W2T = (_Float16*)alloc(1024L * 128 * 2);
  _Float16* H3  = H1;

  auto gemm = [&](const _Float16* A, int lda, const _Float16* Bm, int ldb,
                  float* Cf, _Float16* Ch, int ldc, const float* bias,
                  const float* resid, int M, int N, int K, int gelu) {
    GemmArgs g{A, lda, Bm, ldb, Cf, Ch, ldc, bias, resid, M, N, K, gelu};
    dim3 grid((unsigned)((N + 63) / 64), (unsigned)((M + 63) / 64));
    gemm_f16_wmma<<<grid, 128, 0, stream>>>(g);
  };
  auto cvtT = [&](const float* in, _Float16* out, int N, int K) {
    const long n = (long)N * K;
    cvt_transpose_k<<<(unsigned)((n + 255) / 256), 256, 0, stream>>>(in, out, N, K);
  };

  // patch embed -> token-major residual stream X
  patch_embed_k<<<dim3(KHW_, KB_), 128, 0, stream>>>(x, pe_w, pe_b, X);

  for (int L = 0; L < 4; ++L) {
    const int base = 3 + L * 16;
    const float* ln1_w = (const float*)d_in[base + 0];
    const float* ln1_b = (const float*)d_in[base + 1];
    const float* wq    = (const float*)d_in[base + 2];
    const float* wk    = (const float*)d_in[base + 3];
    const float* wv    = (const float*)d_in[base + 4];
    const float* wo    = (const float*)d_in[base + 5];
    const float* ln2_w = (const float*)d_in[base + 6];
    const float* ln2_b = (const float*)d_in[base + 7];
    const float* w1    = (const float*)d_in[base + 8];
    const float* b1    = (const float*)d_in[base + 9];
    const float* wdw   = (const float*)d_in[base + 10];
    const float* bdw   = (const float*)d_in[base + 11];
    const float* wpw   = (const float*)d_in[base + 12];
    const float* bpw   = (const float*)d_in[base + 13];
    const float* w2    = (const float*)d_in[base + 14];
    const float* b2    = (const float*)d_in[base + 15];

    // ---- SRA attention ----
    layernorm128_k<<<(unsigned)KTOK_, 128, 0, stream>>>(X, ln1_w, ln1_b, XLN);
    cvtT(wq, WQT, 128, 128);
    gemm(XLN, 128, WQT, 128, nullptr, Q, 128, nullptr, nullptr, (int)KTOK_, 128, 128, 0);
    im2col_kv_k<<<(unsigned)(((long)KB_ * KKV_ * 256 + 255) / 256), 256, 0, stream>>>(XLN, COL);
    repack_kv_wT_k<<<(unsigned)((2048L * 128 + 255) / 256), 256, 0, stream>>>(wk, WKT);
    repack_kv_wT_k<<<(unsigned)((2048L * 128 + 255) / 256), 256, 0, stream>>>(wv, WVT);
    gemm(COL, 2048, WKT, 128, nullptr, KT, 128, nullptr, nullptr, KB_ * KKV_, 128, 2048, 0);
    gemm(COL, 2048, WVT, 128, nullptr, VT, 128, nullptr, nullptr, KB_ * KKV_, 128, 2048, 0);
    transpose_kv_k<<<(unsigned)(((long)KB_ * KC_ * KKV_ + 255) / 256), 256, 0, stream>>>(KT, KTt);
    for (int b = 0; b < KB_; ++b) {
      for (int h = 0; h < KHEADS_; ++h) {
        const _Float16* Qbh = Q   + (long)b * KHW_ * KC_ + h * KDIM_;
        const _Float16* Kbh = KTt + (long)b * KC_ * KKV_ + (long)h * KDIM_ * KKV_;
        const _Float16* Vbh = VT  + (long)b * KKV_ * KC_ + h * KDIM_;
        _Float16*       Obh = AO  + (long)b * KHW_ * KC_ + h * KDIM_;
        // S = Q * K^T ; B = KTt slice [K=d][N=j], ldb = 400
        gemm(Qbh, 128, Kbh, KKV_, Sb, nullptr, KKV_, nullptr, nullptr, KHW_, KKV_, KDIM_, 0);
        softmax_rows_k<<<KHW_ / 8, 256, 0, stream>>>(Sb, Pb, KHW_, KKV_, 0.125f);
        // O = P * V ; B = V token-major [K=j][N=d], ldb = 128
        gemm(Pb, KKV_, Vbh, 128, nullptr, Obh, 128, nullptr, nullptr, KHW_, KDIM_, KKV_, 0);
      }
    }
    cvtT(wo, WOT, 128, 128);
    gemm(AO, 128, WOT, 128, X, nullptr, 128, nullptr, X, (int)KTOK_, 128, 128, 0);

    // ---- MixFFN ----
    layernorm128_k<<<(unsigned)KTOK_, 128, 0, stream>>>(X, ln2_w, ln2_b, XLN);
    cvtT(w1, W1T, 1024, 128);
    gemm(XLN, 128, W1T, 1024, nullptr, H1, 1024, b1, nullptr, (int)KTOK_, 1024, 128, 0);
    dwconv3x3_k<<<dim3(KHW_, KB_), 128, 0, stream>>>(H1, wdw, bdw, H2);
    cvtT(wpw, WPT, 1024, 1024);
    gemm(H2, 1024, WPT, 1024, nullptr, H3, 1024, bpw, nullptr, (int)KTOK_, 1024, 1024, 1);
    cvtT(w2, W2T, 128, 1024);
    gemm(H3, 1024, W2T, 128, X, nullptr, 128, b2, X, (int)KTOK_, 128, 1024, 0);
  }

  const long ntot = (long)KB_ * KC_ * KHW_;
  tok_to_nchw_k<<<(unsigned)((ntot + 255) / 256), 256, 0, stream>>>(X, (float*)d_out, ntot);
}
